// BitLinearTrain_28939489640538
// MI455X (gfx1250) — compile-verified, matrix operand order checked by reading, not run on hardware
//
#include <hip/hip_runtime.h>
#include <hip/hip_bf16.h>

// ---- problem dims (fixed by reference) ----
#define BD 16384   // batch rows of X / Y
#define ND 2048    // inner dim (K)
#define MD 2048    // output cols / rows of W

typedef __attribute__((ext_vector_type(16))) __bf16 v16bf;
typedef __attribute__((ext_vector_type(8)))  __bf16 v8bf;
typedef __attribute__((ext_vector_type(4)))  __bf16 v4bf;
typedef __attribute__((ext_vector_type(8)))  float  v8f;

// ------------------------------------------------------------------
// Kernel 0: zero the absmax slot in workspace
// ------------------------------------------------------------------
__global__ void blt_init_ws(unsigned* scaleBits) {
    if (threadIdx.x == 0) *scaleBits = 0u;
}

// ------------------------------------------------------------------
// Kernel 1: scale = max|W|  (positive floats compare correctly as u32)
// ------------------------------------------------------------------
__global__ void blt_absmax(const float* __restrict__ W,
                           unsigned* __restrict__ scaleBits, int n) {
    int tid    = blockIdx.x * blockDim.x + threadIdx.x;
    int stride = gridDim.x * blockDim.x;
    float m = 0.0f;
    for (int i = tid; i < n; i += stride) m = fmaxf(m, fabsf(W[i]));
    for (int off = 16; off > 0; off >>= 1)
        m = fmaxf(m, __shfl_down(m, off, 32));
    __shared__ float smax[8];
    int lane = threadIdx.x & 31, wid = threadIdx.x >> 5;
    if (lane == 0) smax[wid] = m;
    __syncthreads();
    if (wid == 0) {
        m = (lane < (int)(blockDim.x >> 5)) ? smax[lane] : 0.0f;
        for (int off = 4; off > 0; off >>= 1)
            m = fmaxf(m, __shfl_down(m, off, 32));
        if (lane == 0) atomicMax(scaleBits, __float_as_uint(m));
    }
}

// ------------------------------------------------------------------
// Kernel 2: Wq = clip(round(W/(s+1e-8)), -1, 1) as bf16 (exact ternary)
// ------------------------------------------------------------------
__global__ void blt_quant(const float* __restrict__ W,
                          __bf16* __restrict__ Wq,
                          const unsigned* __restrict__ scaleBits, int n) {
    float s = __uint_as_float(*scaleBits) + 1e-8f;
    int tid    = blockIdx.x * blockDim.x + threadIdx.x;
    int stride = gridDim.x * blockDim.x;
    for (int i = tid; i < n; i += stride) {
        float r = rintf(W[i] / s);
        r = fminf(1.0f, fmaxf(-1.0f, r));
        Wq[i] = (__bf16)r;
    }
}

// ------------------------------------------------------------------
// Kernel 2b: one-shot X fp32 -> bf16 (removes cvt chain from GEMM loop)
// ------------------------------------------------------------------
__global__ void blt_cvt_x(const float* __restrict__ X,
                          __bf16* __restrict__ Xb, int n4) {
    int tid    = blockIdx.x * blockDim.x + threadIdx.x;
    int stride = gridDim.x * blockDim.x;
    for (int i = tid; i < n4; i += stride) {
        float4 f = ((const float4*)X)[i];
        v4bf o;
        o[0] = (__bf16)f.x; o[1] = (__bf16)f.y;
        o[2] = (__bf16)f.z; o[3] = (__bf16)f.w;
        ((v4bf*)Xb)[i] = o;
    }
}

// ------------------------------------------------------------------
// Shared epilogue: Y = s*acc + bias (C layout: VGPR r -> M=r / M=r+8)
// ------------------------------------------------------------------
__device__ __forceinline__ void blt_epilogue(
        v8f acc[2][4], const float* __restrict__ bias,
        const unsigned* __restrict__ scaleBits, float* __restrict__ Y,
        int rowBase, int colBase, int l16, int lhalf) {
    const float s = __uint_as_float(*scaleBits);
#pragma unroll
    for (int nt = 0; nt < 4; ++nt) {
        const int col = colBase + nt * 16 + l16;
        const float bv = bias[col];
#pragma unroll
        for (int mt = 0; mt < 2; ++mt) {
            const int row0 = rowBase + mt * 16 + lhalf * 8;
#pragma unroll
            for (int r = 0; r < 8; ++r) {
                float v = s * acc[mt][nt][r] + bv;
                __builtin_nontemporal_store(v, Y + (size_t)(row0 + r) * MD + col);
            }
        }
    }
}

// ------------------------------------------------------------------
// Kernel 3a (preferred): all-bf16 GEMM.  8 waves/block, 128x128 tile,
// wave = 32x64 -> 12 b128 loads + 8 v_wmma per K=32 step, no VALU cvt.
// ------------------------------------------------------------------
__launch_bounds__(256)
__global__ void blt_gemm_xbf(const __bf16* __restrict__ Xb,
                             const __bf16* __restrict__ Wq,
                             const float* __restrict__ bias,
                             const unsigned* __restrict__ scaleBits,
                             float* __restrict__ Y) {
    const int lane  = threadIdx.x & 31;
    const int wid   = threadIdx.x >> 5;
    const int waveM = wid >> 1;           // 0..3
    const int waveN = wid & 1;            // 0..1
    const int lhalf = lane >> 4;          // K sub-block select
    const int l16   = lane & 15;          // M (A) / N (B) row index

    const int rowBase = blockIdx.y * 128 + waveM * 32;
    const int colBase = blockIdx.x * 128 + waveN * 64;

    const __bf16* xRow[2];
    const __bf16* wRow[4];
#pragma unroll
    for (int mt = 0; mt < 2; ++mt)
        xRow[mt] = Xb + (size_t)(rowBase + mt * 16 + l16) * ND;
#pragma unroll
    for (int nt = 0; nt < 4; ++nt)
        wRow[nt] = Wq + (size_t)(colBase + nt * 16 + l16) * ND;

    v8f acc[2][4];
#pragma unroll
    for (int mt = 0; mt < 2; ++mt)
#pragma unroll
        for (int nt = 0; nt < 4; ++nt)
            acc[mt][nt] = (v8f){0.f, 0.f, 0.f, 0.f, 0.f, 0.f, 0.f, 0.f};

    // ISA 7.12.2 bf16 frag per lane: K runs [kb,kb+8) and [kb+16,kb+24),
    // kb = kk + (lane>=16 ? 8 : 0); both operands K-contiguous in memory.
#pragma unroll 2
    for (int kk = 0; kk < ND; kk += 32) {
        const int kb = kk + lhalf * 8;

        v16bf a[2];
#pragma unroll
        for (int mt = 0; mt < 2; ++mt) {
            const __bf16* p = xRow[mt] + kb;
            union { v16bf v; v8bf h[2]; } ua;
            ua.h[0] = *(const v8bf*)(p);
            ua.h[1] = *(const v8bf*)(p + 16);
            a[mt] = ua.v;
        }

        v16bf b[4];
#pragma unroll
        for (int nt = 0; nt < 4; ++nt) {
            const __bf16* p = wRow[nt] + kb;
            union { v16bf v; v8bf h[2]; } ub;
            ub.h[0] = *(const v8bf*)(p);
            ub.h[1] = *(const v8bf*)(p + 16);
            b[nt] = ub.v;
        }

#pragma unroll
        for (int mt = 0; mt < 2; ++mt)
#pragma unroll
            for (int nt = 0; nt < 4; ++nt)
                acc[mt][nt] = __builtin_amdgcn_wmma_f32_16x16x32_bf16(
                    false, a[mt], false, b[nt],
                    (short)0, acc[mt][nt], false, false);
    }

    blt_epilogue(acc, bias, scaleBits, Y, rowBase, colBase, l16, lhalf);
}

// ------------------------------------------------------------------
// Kernel 3b (fallback, small ws): fp32-X GEMM with in-register cvt
// ------------------------------------------------------------------
__launch_bounds__(256)
__global__ void blt_gemm_xf32(const float* __restrict__ X,
                              const __bf16* __restrict__ Wq,
                              const float* __restrict__ bias,
                              const unsigned* __restrict__ scaleBits,
                              float* __restrict__ Y) {
    const int lane  = threadIdx.x & 31;
    const int wid   = threadIdx.x >> 5;
    const int waveM = wid >> 1;
    const int waveN = wid & 1;
    const int lhalf = lane >> 4;
    const int l16   = lane & 15;

    const int rowBase = blockIdx.y * 128 + waveM * 32;
    const int colBase = blockIdx.x * 128 + waveN * 64;

    const float*  xRow[2];
    const __bf16* wRow[4];
#pragma unroll
    for (int mt = 0; mt < 2; ++mt)
        xRow[mt] = X + (size_t)(rowBase + mt * 16 + l16) * ND;
#pragma unroll
    for (int nt = 0; nt < 4; ++nt)
        wRow[nt] = Wq + (size_t)(colBase + nt * 16 + l16) * ND;

    v8f acc[2][4];
#pragma unroll
    for (int mt = 0; mt < 2; ++mt)
#pragma unroll
        for (int nt = 0; nt < 4; ++nt)
            acc[mt][nt] = (v8f){0.f, 0.f, 0.f, 0.f, 0.f, 0.f, 0.f, 0.f};

    for (int kk = 0; kk < ND; kk += 32) {
        const int kb = kk + lhalf * 8;

        v16bf a[2];
#pragma unroll
        for (int mt = 0; mt < 2; ++mt) {
            const float* p = xRow[mt] + kb;
            float4 f0 = *(const float4*)(p);
            float4 f1 = *(const float4*)(p + 4);
            float4 f2 = *(const float4*)(p + 16);
            float4 f3 = *(const float4*)(p + 20);
            union { v16bf v; __bf16 e[16]; } ua;
            ua.e[0]  = (__bf16)f0.x; ua.e[1]  = (__bf16)f0.y;
            ua.e[2]  = (__bf16)f0.z; ua.e[3]  = (__bf16)f0.w;
            ua.e[4]  = (__bf16)f1.x; ua.e[5]  = (__bf16)f1.y;
            ua.e[6]  = (__bf16)f1.z; ua.e[7]  = (__bf16)f1.w;
            ua.e[8]  = (__bf16)f2.x; ua.e[9]  = (__bf16)f2.y;
            ua.e[10] = (__bf16)f2.z; ua.e[11] = (__bf16)f2.w;
            ua.e[12] = (__bf16)f3.x; ua.e[13] = (__bf16)f3.y;
            ua.e[14] = (__bf16)f3.z; ua.e[15] = (__bf16)f3.w;
            a[mt] = ua.v;
        }

        v16bf b[4];
#pragma unroll
        for (int nt = 0; nt < 4; ++nt) {
            const __bf16* p = wRow[nt] + kb;
            union { v16bf v; v8bf h[2]; } ub;
            ub.h[0] = *(const v8bf*)(p);
            ub.h[1] = *(const v8bf*)(p + 16);
            b[nt] = ub.v;
        }

#pragma unroll
        for (int mt = 0; mt < 2; ++mt)
#pragma unroll
            for (int nt = 0; nt < 4; ++nt)
                acc[mt][nt] = __builtin_amdgcn_wmma_f32_16x16x32_bf16(
                    false, a[mt], false, b[nt],
                    (short)0, acc[mt][nt], false, false);
    }

    blt_epilogue(acc, bias, scaleBits, Y, rowBase, colBase, l16, lhalf);
}

// ------------------------------------------------------------------
extern "C" void kernel_launch(void* const* d_in, const int* in_sizes, int n_in,
                              void* d_out, int out_size, void* d_ws, size_t ws_size,
                              hipStream_t stream) {
    const float* X    = (const float*)d_in[0];
    const float* W    = (const float*)d_in[1];
    const float* bias = (const float*)d_in[2];
    float* Y          = (float*)d_out;

    // ws layout: [0,4): scale bits; [256,+8MB): Wq bf16; [+8MB,+64MB): X bf16
    const size_t wqBytes = (size_t)MD * ND * sizeof(__bf16);   //  8 MB
    const size_t xbBytes = (size_t)BD * ND * sizeof(__bf16);   // 64 MB
    unsigned* scaleBits = (unsigned*)d_ws;
    __bf16*   Wq        = (__bf16*)((char*)d_ws + 256);
    __bf16*   Xb        = (__bf16*)((char*)d_ws + 256 + wqBytes);

    const int nW = MD * ND;

    blt_init_ws<<<1, 64, 0, stream>>>(scaleBits);
    blt_absmax<<<1024, 256, 0, stream>>>(W, scaleBits, nW);
    blt_quant<<<2048, 256, 0, stream>>>(W, Wq, scaleBits, nW);

    dim3 grid(MD / 128, BD / 128);   // (16, 128): N-tiles fast for X L2 reuse
    if (ws_size >= 256 + wqBytes + xbBytes) {
        blt_cvt_x<<<4096, 256, 0, stream>>>(X, Xb, BD * ND / 4);
        blt_gemm_xbf<<<grid, 256, 0, stream>>>(Xb, Wq, bias, scaleBits, Y);
    } else {
        blt_gemm_xf32<<<grid, 256, 0, stream>>>(X, Wq, bias, scaleBits, Y);
    }
}
